// BNLSTM_5102421147914
// MI455X (gfx1250) — matrix operand order repaired and checked by reading, not compiled, not run on hardware
//
#include <hip/hip_runtime.h>
#include <stdint.h>

// ---------------- CDNA5 WMMA fragment types ----------------
typedef __attribute__((ext_vector_type(16))) unsigned short v16u;
typedef __attribute__((ext_vector_type(8)))  unsigned short v8u;
typedef __attribute__((ext_vector_type(16))) __bf16         v16bf;
typedef __attribute__((ext_vector_type(8)))  float          v8f;

#define B_DIM   2048
#define T_DIM   152
#define I_DIM   75
#define IPAD    96       // 75 padded to 3 K-tiles of 32
#define H_DIM   128
#define G_DIM   512      // 4H
#define O_DIM   256
#define NWG     8        // 8 workgroups, each owns 16 hidden units (64 gate cols)
#define NTHREADS 512     // 16 waves -> 4 waves/SIMD -> 128 VGPRs/wave
#define NWAVES  16
#define MT_TOTAL 128     // 2048 / 16 row tiles
#define MT_PER_WAVE 8    // 128 tiles / 16 waves
#define EPSF    1e-5f

static __device__ __forceinline__ unsigned short f2bf(float f) {
  union { float f; unsigned u; } x; x.f = f;
  unsigned u = x.u;
  return (unsigned short)((u + 0x7FFFu + ((u >> 16) & 1u)) >> 16);
}
static __device__ __forceinline__ float sigmoidf_(float x) {
  return 1.f / (1.f + __expf(-x));
}
static __device__ __forceinline__ float tanhf_(float x) {
  x = fminf(fmaxf(x, -15.f), 15.f);
  float e = __expf(2.f * x);
  return (e - 1.f) / (e + 1.f);
}
static __device__ __forceinline__ v8f wmma_bf16(v16u a, v16u b, v8f c) {
  return __builtin_amdgcn_wmma_f32_16x16x32_bf16(
      false, __builtin_bit_cast(v16bf, a),
      false, __builtin_bit_cast(v16bf, b),
      (short)0, c, false, false);
}

// A fragment (16x32 bf16) from a row-major bf16 matrix.
// Contiguous-k order used here (not AMD's table order): lane L -> row m0+(L&15); half h -> k = h + (L>>4)*16,
// i.e. two contiguous 16-byte runs per lane -> two b128 loads, no repacking ALU.
static __device__ __forceinline__ v16u a_frag_bf16(const unsigned short* __restrict__ base,
                                                   int m0, long rowStride, int k0) {
  int lane = threadIdx.x & 31;
  const unsigned short* row = base + (long)(m0 + (lane & 15)) * rowStride
                                   + k0 + ((lane >> 4) << 4);
  v8u lo = *(const v8u*)(row);
  v8u hi = *(const v8u*)(row + 8);
  return __builtin_shufflevector(lo, hi, 0, 1, 2, 3, 4, 5, 6, 7,
                                 8, 9, 10, 11, 12, 13, 14, 15);
}

// ---------------- persistent BN-LSTM layer kernel ----------------
// grid = 8 WGs x 512 threads; WG wg owns hidden units [wg*16, wg*16+16),
// i.e. gate columns {g*128 + wg*16 + c} for gates g = f,i,o,g.
__global__ __launch_bounds__(NTHREADS, 1)
void bnlstm_layer(const unsigned short* __restrict__ xin,  // bf16 input
                  long x_tmul,        // element offset per timestep
                  long x_rstride,     // element stride per batch row
                  int K_in, int kt_in,
                  const float* __restrict__ w_ih,   // (K_in, 512) row-major f32
                  const float* __restrict__ w_hh,   // (128, 512)  row-major f32
                  const float* __restrict__ bias,
                  const float* __restrict__ g_ih, const float* __restrict__ be_ih,
                  const float* __restrict__ g_hh, const float* __restrict__ be_hh,
                  const float* __restrict__ g_c,  const float* __restrict__ be_c,
                  unsigned short* __restrict__ h_state,  // (2048,128) bf16, zero-init
                  float* __restrict__ c_state,           // (2048,128) f32, zero-init
                  float* __restrict__ ws_wh,             // spill: wh preacts / sigma(o)
                  float* __restrict__ ws_wi,             // spill: wi preacts
                  unsigned short* __restrict__ out_bf16, // layer0: (T,2048,128) or null
                  unsigned int* __restrict__ bar) {
  // Pre-swizzled weight B-fragments: frag = kt*4+nt, per-lane 16 halves contiguous.
  __shared__ __attribute__((aligned(32))) unsigned short lds_whh[16 * 512];
  __shared__ __attribute__((aligned(32))) unsigned short lds_wih[16 * 512];
  __shared__ float s_sum_h[64], s_sq_h[64], s_sum_i[64], s_sq_i[64];
  __shared__ float s_Ah[64], s_Ai[64], s_K0[64];
  __shared__ float s_csum[16], s_csq[16], s_Ac[16], s_Kc[16];

  const int wg   = blockIdx.x;
  const int j0   = wg * 16;
  const int tid  = threadIdx.x;
  const int wave = tid >> 5;
  const int lane = tid & 31;
  const int ln   = lane & 15;   // column within 16-wide tile
  const int grp  = lane >> 4;

  // ---- one-time: swizzle this WG's weight slice into LDS fragment layout.
  // B-layout: lane L -> col n = nt*16 + (L&15); half h -> k = kt*32 + (L>>4)*16 + h.
  for (int idx = tid; idx < 16 * 512; idx += NTHREADS) {
    int frag = idx >> 9, rem = idx & 511;
    int l = rem >> 4, h = rem & 15;
    int kt = frag >> 2, nt = frag & 3;
    int k = kt * 32 + ((l >> 4) << 4) + h;
    int gcol = nt * H_DIM + j0 + (l & 15);
    lds_whh[idx] = f2bf(w_hh[k * G_DIM + gcol]);            // k < 128 always
    lds_wih[idx] = (k < K_in) ? f2bf(w_ih[k * G_DIM + gcol]) : (unsigned short)0;
  }
  __syncthreads();

#pragma unroll 1
  for (int t = 0; t < T_DIM; ++t) {
    if (tid < 64) { s_sum_h[tid] = 0.f; s_sq_h[tid] = 0.f;
                    s_sum_i[tid] = 0.f; s_sq_i[tid] = 0.f; }
    if (tid < 16) { s_csum[tid] = 0.f; s_csq[tid] = 0.f; }
    __syncthreads();

    const unsigned short* xin_t = xin + (long)t * x_tmul;

    // ================= Phase A: GEMMs + column stats =================
    float psum_h[4] = {0,0,0,0}, psq_h[4] = {0,0,0,0};
    float psum_i[4] = {0,0,0,0}, psq_i[4] = {0,0,0,0};
#pragma unroll 1
    for (int mi = 0; mi < MT_PER_WAVE; ++mi) {
      const int mt = wave * MT_PER_WAVE + mi;
      const int m0 = mt * 16;
      // --- wh = h @ w_hh (K = 128)
      {
        v8f acc[4] = {v8f{}, v8f{}, v8f{}, v8f{}};
#pragma unroll
        for (int kt = 0; kt < 4; ++kt) {
          v16u a = a_frag_bf16(h_state, m0, H_DIM, kt * 32);
#pragma unroll
          for (int nt = 0; nt < 4; ++nt) {
            v16u b = *(const v16u*)&lds_whh[(kt * 4 + nt) * 512 + lane * 16];
            acc[nt] = wmma_bf16(a, b, acc[nt]);
          }
        }
#pragma unroll
        for (int nt = 0; nt < 4; ++nt) {
          float s = 0.f, q = 0.f;
#pragma unroll
          for (int r = 0; r < 8; ++r) { float v = acc[nt][r]; s += v; q += v * v; }
          psum_h[nt] += s; psq_h[nt] += q;
          float* dst = ws_wh + ((((long)wg * MT_TOTAL + mt) * 4 + nt) << 8) + lane * 8;
          *(v8f*)dst = acc[nt];
        }
      }
      // --- wi = x_t @ w_ih (K = 96(pad) or 128)
      {
        v8f acc[4] = {v8f{}, v8f{}, v8f{}, v8f{}};
#pragma unroll 1
        for (int kt = 0; kt < kt_in; ++kt) {
          v16u a = a_frag_bf16(xin_t, m0, x_rstride, kt * 32);
#pragma unroll
          for (int nt = 0; nt < 4; ++nt) {
            v16u b = *(const v16u*)&lds_wih[(kt * 4 + nt) * 512 + lane * 16];
            acc[nt] = wmma_bf16(a, b, acc[nt]);
          }
        }
#pragma unroll
        for (int nt = 0; nt < 4; ++nt) {
          float s = 0.f, q = 0.f;
#pragma unroll
          for (int r = 0; r < 8; ++r) { float v = acc[nt][r]; s += v; q += v * v; }
          psum_i[nt] += s; psq_i[nt] += q;
          float* dst = ws_wi + ((((long)wg * MT_TOTAL + mt) * 4 + nt) << 8) + lane * 8;
          *(v8f*)dst = acc[nt];
        }
      }
    }
#pragma unroll
    for (int nt = 0; nt < 4; ++nt) {
      atomicAdd(&s_sum_h[nt * 16 + ln], psum_h[nt]);
      atomicAdd(&s_sq_h [nt * 16 + ln], psq_h [nt]);
      atomicAdd(&s_sum_i[nt * 16 + ln], psum_i[nt]);
      atomicAdd(&s_sq_i [nt * 16 + ln], psq_i [nt]);
    }
    __syncthreads();

    // fold the two BNs + bias into one affine per gate column
    if (tid < 64) {
      int nt = tid >> 4, c = tid & 15;
      int gcol = nt * H_DIM + j0 + c;
      const float inv = 1.f / (float)B_DIM;
      float mh = s_sum_h[tid] * inv;
      float vh = s_sq_h[tid] * inv - mh * mh;
      float Ah = g_hh[gcol] * rsqrtf(vh + EPSF);
      float mi2 = s_sum_i[tid] * inv;
      float vi = s_sq_i[tid] * inv - mi2 * mi2;
      float Ai = g_ih[gcol] * rsqrtf(vi + EPSF);
      s_Ah[tid] = Ah; s_Ai[tid] = Ai;
      s_K0[tid] = be_hh[gcol] + be_ih[gcol] + bias[gcol] - mh * Ah - mi2 * Ai;
    }
    __syncthreads();

    // ================= Phase B: gates + c update + c stats =================
    float pcs = 0.f, pcq = 0.f;
    const int jc = j0 + ln;
#pragma unroll 1
    for (int mi = 0; mi < MT_PER_WAVE; ++mi) {
      const int mt = wave * MT_PER_WAVE + mi;
      v8f g4[4];
#pragma unroll
      for (int nt = 0; nt < 4; ++nt) {
        long base = (((long)wg * MT_TOTAL + mt) * 4 + nt) << 8;
        v8f wh = *(const v8f*)(ws_wh + base + lane * 8);
        v8f wi = *(const v8f*)(ws_wi + base + lane * 8);
        float A1 = s_Ah[nt * 16 + ln], A2 = s_Ai[nt * 16 + ln], K = s_K0[nt * 16 + ln];
#pragma unroll
        for (int r = 0; r < 8; ++r) g4[nt][r] = A1 * wh[r] + A2 * wi[r] + K;
      }
      const int mbase = mt * 16 + grp * 8;
      v8f so;
#pragma unroll
      for (int r = 0; r < 8; ++r) {
        long ci = (long)(mbase + r) * H_DIM + jc;
        float cold = c_state[ci];
        float fg = sigmoidf_(g4[0][r]);
        float ig = sigmoidf_(g4[1][r]);
        float og = sigmoidf_(g4[2][r]);
        float gg = tanhf_(g4[3][r]);
        float c1 = fg * cold + ig * gg;
        c_state[ci] = c1;
        pcs += c1; pcq += c1 * c1;
        so[r] = og;
      }
      // stash sigma(o) in the dead wh spill slot (nt=0) for Phase C
      float* po = ws_wh + (((long)wg * MT_TOTAL + mt) * 4 << 8) + lane * 8;
      *(v8f*)po = so;
    }
    atomicAdd(&s_csum[ln], pcs);
    atomicAdd(&s_csq [ln], pcq);
    __syncthreads();

    if (tid < 16) {
      int j = j0 + tid;
      const float inv = 1.f / (float)B_DIM;
      float mc = s_csum[tid] * inv;
      float vc = s_csq[tid] * inv - mc * mc;
      float Ac = g_c[j] * rsqrtf(vc + EPSF);
      s_Ac[tid] = Ac; s_Kc[tid] = be_c[j] - mc * Ac;
    }
    __syncthreads();

    // ================= Phase C: h = sigma(o)*tanh(BN(c)) =================
    {
      float Ac = s_Ac[ln], Kc = s_Kc[ln];
#pragma unroll 1
      for (int mi = 0; mi < MT_PER_WAVE; ++mi) {
        const int mt = wave * MT_PER_WAVE + mi;
        const float* po = ws_wh + (((long)wg * MT_TOTAL + mt) * 4 << 8) + lane * 8;
        v8f so = *(const v8f*)po;
        const int mbase = mt * 16 + grp * 8;
#pragma unroll
        for (int r = 0; r < 8; ++r) {
          int m = mbase + r;
          float c1 = c_state[(long)m * H_DIM + jc];
          float hv = so[r] * tanhf_(Ac * c1 + Kc);
          unsigned short hb = f2bf(hv);
          h_state[(long)m * H_DIM + jc] = hb;
          if (out_bf16) out_bf16[((long)t * B_DIM + m) * H_DIM + jc] = hb;
        }
      }
    }

    // ================= device barrier across the 8 WGs =================
    __threadfence();
    __syncthreads();
    if (tid == 0) {
      __hip_atomic_fetch_add(bar, 1u, __ATOMIC_RELEASE, __HIP_MEMORY_SCOPE_AGENT);
      unsigned tgt = (unsigned)(t + 1) * NWG;
      while (__hip_atomic_load(bar, __ATOMIC_ACQUIRE, __HIP_MEMORY_SCOPE_AGENT) < tgt)
        __builtin_amdgcn_s_sleep(2);
    }
    __syncthreads();
    __builtin_amdgcn_fence(__ATOMIC_ACQUIRE, "agent");
  }
}

// ---------------- pre-convert sequences (B,T,75) f32 -> (B,T,96) bf16, zero-padded
__global__ void xconv_kernel(const float* __restrict__ seq,
                             unsigned short* __restrict__ xpad) {
  long i = (long)blockIdx.x * blockDim.x + threadIdx.x;
  long n = (long)B_DIM * T_DIM * IPAD;
  if (i >= n) return;
  int k = (int)(i % IPAD);
  long bt = i / IPAD;
  xpad[i] = (k < I_DIM) ? f2bf(seq[bt * I_DIM + k]) : (unsigned short)0;
}

// ---------------- FC: out(2048x256) = hlast(2048x128) @ fc_w^T + b ----------------
__global__ __launch_bounds__(256)
void fc_kernel(const unsigned short* __restrict__ hlast, const float* __restrict__ fc_w,
               const float* __restrict__ fc_b, float* __restrict__ out) {
  int wid = (blockIdx.x * 256 + threadIdx.x) >> 5;   // 2048 wave-tiles
  int lane = threadIdx.x & 31;
  int mt = wid >> 4, nt = wid & 15;
  int m0 = mt * 16;
  v8f acc = {};
#pragma unroll
  for (int kt = 0; kt < 4; ++kt) {
    v16u a = a_frag_bf16(hlast, m0, H_DIM, kt * 32);
    v16u b;
    {
      int n = nt * 16 + (lane & 15);
      const float* row = fc_w + (long)n * H_DIM + kt * 32 + ((lane >> 4) << 4);
#pragma unroll
      for (int h = 0; h < 16; ++h) b[h] = f2bf(row[h]);  // B[k][n] = fc_w[n][k]
    }
    acc = wmma_bf16(a, b, acc);
  }
  int n = nt * 16 + (lane & 15);
  float bv = fc_b[n];
  int mb = m0 + ((lane >> 4) << 3);
#pragma unroll
  for (int r = 0; r < 8; ++r) out[(long)(mb + r) * O_DIM + n] = acc[r] + bv;
}

__global__ void init_kernel(unsigned int* h0, float* c0, unsigned int* h1, float* c1,
                            unsigned int* bars) {
  long i = (long)blockIdx.x * blockDim.x + threadIdx.x;
  if (i < (long)B_DIM * H_DIM) { c0[i] = 0.f; c1[i] = 0.f; }
  if (i < (long)B_DIM * H_DIM / 2) { h0[i] = 0u; h1[i] = 0u; }  // bf16 pairs
  if (i < 8) bars[i] = 0u;
}

extern "C" void kernel_launch(void* const* d_in, const int* in_sizes, int n_in,
                              void* d_out, int out_size, void* d_ws, size_t ws_size,
                              hipStream_t stream) {
  (void)in_sizes; (void)n_in; (void)out_size; (void)ws_size;
  const float* seq    = (const float*)d_in[0];
  const float* w_ih0  = (const float*)d_in[1];
  const float* w_hh0  = (const float*)d_in[2];
  const float* b0     = (const float*)d_in[3];
  const float* g_ih0  = (const float*)d_in[4];
  const float* be_ih0 = (const float*)d_in[5];
  const float* g_hh0  = (const float*)d_in[6];
  const float* be_hh0 = (const float*)d_in[7];
  const float* g_c0   = (const float*)d_in[8];
  const float* be_c0  = (const float*)d_in[9];
  const float* w_ih1  = (const float*)d_in[10];
  const float* w_hh1  = (const float*)d_in[11];
  const float* b1     = (const float*)d_in[12];
  const float* g_ih1  = (const float*)d_in[13];
  const float* be_ih1 = (const float*)d_in[14];
  const float* g_hh1  = (const float*)d_in[15];
  const float* be_hh1 = (const float*)d_in[16];
  const float* g_c1   = (const float*)d_in[17];
  const float* be_c1  = (const float*)d_in[18];
  const float* fc_w   = (const float*)d_in[19];
  const float* fc_b   = (const float*)d_in[20];

  char* ws = (char*)d_ws;
  size_t off = 0;
  auto alloc = [&](size_t bytes) -> void* {
    void* p = ws + off;
    off = (off + bytes + 255) & ~(size_t)255;
    return p;
  };
  unsigned int* bars = (unsigned int*)alloc(256);
  unsigned short* h0 = (unsigned short*)alloc((size_t)B_DIM * H_DIM * 2);
  unsigned short* h1 = (unsigned short*)alloc((size_t)B_DIM * H_DIM * 2);
  float* c0 = (float*)alloc((size_t)B_DIM * H_DIM * 4);
  float* c1 = (float*)alloc((size_t)B_DIM * H_DIM * 4);
  float* ws_wh = (float*)alloc((size_t)NWG * MT_TOTAL * 4 * 256 * 4);   // 4 MB
  float* ws_wi = (float*)alloc((size_t)NWG * MT_TOTAL * 4 * 256 * 4);   // 4 MB
  unsigned short* xpad =
      (unsigned short*)alloc((size_t)B_DIM * T_DIM * IPAD * 2);         // ~57 MB
  unsigned short* out0 =
      (unsigned short*)alloc((size_t)T_DIM * B_DIM * H_DIM * 2);        // ~80 MB

  init_kernel<<<(B_DIM * H_DIM + 255) / 256, 256, 0, stream>>>(
      (unsigned int*)h0, c0, (unsigned int*)h1, c1, bars);

  {
    long n = (long)B_DIM * T_DIM * IPAD;
    xconv_kernel<<<(int)((n + 255) / 256), 256, 0, stream>>>(seq, xpad);
  }

  // layer 0: x from (B,T,96) bf16  -> per-t base offset 96, row stride T*96
  bnlstm_layer<<<NWG, NTHREADS, 0, stream>>>(
      xpad, (long)IPAD, (long)T_DIM * IPAD, I_DIM, 3,
      w_ih0, w_hh0, b0, g_ih0, be_ih0, g_hh0, be_hh0, g_c0, be_c0,
      h0, c0, ws_wh, ws_wi, out0, &bars[0]);

  // layer 1: x from (T,B,128) bf16 -> per-t base offset B*H, row stride 128
  bnlstm_layer<<<NWG, NTHREADS, 0, stream>>>(
      out0, (long)B_DIM * H_DIM, (long)H_DIM, H_DIM, 4,
      w_ih1, w_hh1, b1, g_ih1, be_ih1, g_hh1, be_hh1, g_c1, be_c1,
      h1, c1, ws_wh, ws_wi, nullptr, &bars[1]);

  fc_kernel<<<256, 256, 0, stream>>>(h1, fc_w, fc_b, (float*)d_out);
}
